// Cmpd_d_MPNN_3917010174549
// MI455X (gfx1250) — matrix operand order repaired and checked by reading, not compiled, not run on hardware
//
#include <hip/hip_runtime.h>
#include <hip/hip_bf16.h>

// ---------------- problem constants (from reference) ----------------
constexpr int N_ATOMS_C = 150001;
constexpr int N_BONDS_C = 300001;
constexpr int MAX_NB_C  = 6;
constexpr int AFD_C     = 133;   // ATOM_FDIM
constexpr int BFD_C     = 147;   // BOND_FDIM
constexpr int H_C       = 300;   // HIDDEN
constexpr int N_MOLS_C  = 10000;

// padded sizes
constexpr int MA_C  = 150016;    // atoms padded to 16
constexpr int MB_C  = 300016;    // bonds padded to 16
constexpr int HB_C  = 320;       // hidden padded to 32 (k-step) ; 20 n-tiles of 16
constexpr int KFB_C = 160;       // bond feature dim padded to 32
constexpr int KAI_C = 448;       // atom_fdim + hidden padded to 32
constexpr int NTILES_C = HB_C / 16;   // 20

typedef __attribute__((ext_vector_type(16))) __bf16 v16bf;
typedef __attribute__((ext_vector_type(8)))  float  v8f;
typedef __attribute__((ext_vector_type(4)))  unsigned int u32x4;

// ---------------- WMMA fragment loaders ----------------
// A (16x32 bf16, row-major source, stride lda):
//   lane<16 : row = m0+lane,    elems 0..15 = K k0..k0+15
//   lane>=16: row = m0+lane-16, elems 0..15 = K k0+16..k0+31
__device__ __forceinline__ v16bf load_a_frag(const __bf16* __restrict__ A, int lda,
                                             int m0, int kt, int lane) {
  const int row = m0 + (lane & 15);
  const int kb  = kt * 32 + ((lane & 16) ? 16 : 0);
  const __bf16* p = A + (size_t)row * lda + kb;
  union { v16bf v; u32x4 u[2]; } x;
  x.u[0] = *(const u32x4*)(p);
  x.u[1] = *(const u32x4*)(p + 8);
  return x.v;
}

// B pre-packed: 32 contiguous bytes per lane per (kt, nt)
__device__ __forceinline__ v16bf load_b_frag(const __bf16* __restrict__ Bp,
                                             int kt, int nt, int lane) {
  const __bf16* p = Bp + (((size_t)(kt * NTILES_C + nt) * 32 + lane) * 16);
  union { v16bf v; u32x4 u[2]; } x;
  x.u[0] = *(const u32x4*)(p);
  x.u[1] = *(const u32x4*)(p + 8);
  return x.v;
}

// ---------------- prep kernels ----------------
// f_bonds fp32 [N_BONDS,147] -> bf16 padded [MB, 160] (zero-padded)
__global__ void conv_bonds_kernel(const float* __restrict__ src, __bf16* __restrict__ dst) {
  int tid = blockIdx.x * blockDim.x + threadIdx.x;
  constexpr int tot = MB_C * KFB_C;
  if (tid >= tot) return;
  int row = tid / KFB_C;
  int col = tid - row * KFB_C;
  float v = 0.f;
  if (row < N_BONDS_C && col < BFD_C) v = src[(size_t)row * BFD_C + col];
  dst[tid] = (__bf16)v;
}

// Pack fp32 weight [Korig, Norig] into WMMA B fragments, padded to [Kpad, 320].
// Layout per (kt, nt, lane): lane<16 -> col=n0+lane, K=k0+j (j=0..15)
//                            lane>=16 -> col=n0+lane-16, K=k0+16+j
__global__ void pack_w_kernel(const float* __restrict__ W, int Korig, int Norig,
                              int Kpad, __bf16* __restrict__ out) {
  int tid = blockIdx.x * blockDim.x + threadIdx.x;
  const int total = (Kpad / 32) * NTILES_C * 32;
  if (tid >= total) return;
  const int lane = tid & 31;
  const int nt = (tid >> 5) % NTILES_C;
  const int kt = (tid >> 5) / NTILES_C;
  const int col = nt * 16 + (lane & 15);
  const int kb  = kt * 32 + ((lane & 16) ? 16 : 0);
  union { u32x4 u[2]; __bf16 h[16]; } x;
#pragma unroll
  for (int j = 0; j < 16; ++j) {
    const int k = kb + j;
    float v = (k < Korig && col < Norig) ? W[(size_t)k * Norig + col] : 0.f;
    x.h[j] = (__bf16)v;
  }
  u32x4* dst = (u32x4*)(out + (size_t)tid * 16);
  dst[0] = x.u[0];
  dst[1] = x.u[1];
}

// ---------------- GEMM kernels (wave = 16 rows x 160 cols, 10 WMMA tiles) ----------------
// GEMM1: inp = fb @ W_i ; msg = relu(inp)   (both stored bf16, stride 320)
__global__ __launch_bounds__(128) void gemm_wi_kernel(
    const __bf16* __restrict__ A, const __bf16* __restrict__ Bp,
    __bf16* __restrict__ inp, __bf16* __restrict__ msg,
    int Mtiles, int Ktiles, int lda)
{
  const int lane = threadIdx.x & 31;
  const int wave = threadIdx.x >> 5;
  const int work = blockIdx.x * 4 + wave;
  const int mt = work >> 1;
  const int nh = work & 1;
  if (mt >= Mtiles) return;
  const int m0 = mt * 16;
  v8f acc[10] = {};
  for (int kt = 0; kt < Ktiles; ++kt) {
    v16bf a = load_a_frag(A, lda, m0, kt, lane);
#pragma unroll
    for (int nt = 0; nt < 10; ++nt) {
      v16bf b = load_b_frag(Bp, kt, nh * 10 + nt, lane);
      acc[nt] = __builtin_amdgcn_wmma_f32_16x16x32_bf16(false, a, false, b,
                                                        (short)0, acc[nt], false, false);
    }
  }
  const int col = lane & 15;
  const int rbase = m0 + ((lane & 16) ? 8 : 0);
#pragma unroll
  for (int nt = 0; nt < 10; ++nt) {
    const int n = (nh * 10 + nt) * 16 + col;
#pragma unroll
    for (int v = 0; v < 8; ++v) {
      const size_t idx = (size_t)(rbase + v) * HB_C + n;
      float x = acc[nt][v];
      inp[idx] = (__bf16)x;
      msg[idx] = (__bf16)fmaxf(x, 0.f);
    }
  }
}

// GEMM2: msg = relu(inp + t @ W_h)
__global__ __launch_bounds__(128) void gemm_wh_kernel(
    const __bf16* __restrict__ A, const __bf16* __restrict__ Bp,
    const __bf16* __restrict__ inp, __bf16* __restrict__ msg,
    int Mtiles, int Ktiles, int lda)
{
  const int lane = threadIdx.x & 31;
  const int wave = threadIdx.x >> 5;
  const int work = blockIdx.x * 4 + wave;
  const int mt = work >> 1;
  const int nh = work & 1;
  if (mt >= Mtiles) return;
  const int m0 = mt * 16;
  v8f acc[10] = {};
  for (int kt = 0; kt < Ktiles; ++kt) {
    v16bf a = load_a_frag(A, lda, m0, kt, lane);
#pragma unroll
    for (int nt = 0; nt < 10; ++nt) {
      v16bf b = load_b_frag(Bp, kt, nh * 10 + nt, lane);
      acc[nt] = __builtin_amdgcn_wmma_f32_16x16x32_bf16(false, a, false, b,
                                                        (short)0, acc[nt], false, false);
    }
  }
  const int col = lane & 15;
  const int rbase = m0 + ((lane & 16) ? 8 : 0);
#pragma unroll
  for (int nt = 0; nt < 10; ++nt) {
    const int n = (nh * 10 + nt) * 16 + col;
#pragma unroll
    for (int v = 0; v < 8; ++v) {
      const size_t idx = (size_t)(rbase + v) * HB_C + n;
      float x = acc[nt][v] + (float)inp[idx];
      msg[idx] = (__bf16)fmaxf(x, 0.f);
    }
  }
}

// GEMM3: ah = relu(ain @ W_o + b_o)   (fp32 out, stride 320)
__global__ __launch_bounds__(128) void gemm_wo_kernel(
    const __bf16* __restrict__ A, const __bf16* __restrict__ Bp,
    const float* __restrict__ b_o, float* __restrict__ ah,
    int Mtiles, int Ktiles, int lda)
{
  const int lane = threadIdx.x & 31;
  const int wave = threadIdx.x >> 5;
  const int work = blockIdx.x * 4 + wave;
  const int mt = work >> 1;
  const int nh = work & 1;
  if (mt >= Mtiles) return;
  const int m0 = mt * 16;
  v8f acc[10] = {};
  for (int kt = 0; kt < Ktiles; ++kt) {
    v16bf a = load_a_frag(A, lda, m0, kt, lane);
#pragma unroll
    for (int nt = 0; nt < 10; ++nt) {
      v16bf b = load_b_frag(Bp, kt, nh * 10 + nt, lane);
      acc[nt] = __builtin_amdgcn_wmma_f32_16x16x32_bf16(false, a, false, b,
                                                        (short)0, acc[nt], false, false);
    }
  }
  const int col = lane & 15;
  const int rbase = m0 + ((lane & 16) ? 8 : 0);
#pragma unroll
  for (int nt = 0; nt < 10; ++nt) {
    const int n = (nh * 10 + nt) * 16 + col;
    const float bias = (n < H_C) ? b_o[n] : 0.f;
#pragma unroll
    for (int v = 0; v < 8; ++v) {
      const size_t idx = (size_t)(rbase + v) * HB_C + n;
      ah[idx] = fmaxf(acc[nt][v] + bias, 0.f);
    }
  }
}

// ---------------- gather / scatter kernels ----------------
// a_message[atom] = sum_k message[a2b[atom][k]]   (bf16 in, fp32 accum, bf16 out)
__global__ void agg_atoms_kernel(const __bf16* __restrict__ msg, const int* __restrict__ a2b,
                                 __bf16* __restrict__ amsg) {
  int tid = blockIdx.x * blockDim.x + threadIdx.x;
  constexpr int tot = MA_C * (HB_C / 8);
  if (tid >= tot) return;
  int atom = tid / (HB_C / 8);
  int h0 = (tid - atom * (HB_C / 8)) * 8;
  float s[8] = {0.f, 0.f, 0.f, 0.f, 0.f, 0.f, 0.f, 0.f};
  if (atom < N_ATOMS_C) {
#pragma unroll
    for (int k = 0; k < MAX_NB_C; ++k) {
      int b = a2b[atom * MAX_NB_C + k];
      union { u32x4 u; __bf16 h[8]; } x;
      x.u = *(const u32x4*)(msg + (size_t)b * HB_C + h0);
#pragma unroll
      for (int j = 0; j < 8; ++j) s[j] += (float)x.h[j];
    }
  }
  union { u32x4 u; __bf16 h[8]; } o;
#pragma unroll
  for (int j = 0; j < 8; ++j) o.h[j] = (__bf16)s[j];
  *(u32x4*)(amsg + (size_t)atom * HB_C + h0) = o.u;
}

// t[b] = amsg[b2a[b]] - msg[b2revb[b]]
__global__ void build_t_kernel(const __bf16* __restrict__ amsg, const __bf16* __restrict__ msg,
                               const int* __restrict__ b2a, const int* __restrict__ b2revb,
                               __bf16* __restrict__ t) {
  int tid = blockIdx.x * blockDim.x + threadIdx.x;
  constexpr int tot = MB_C * (HB_C / 8);
  if (tid >= tot) return;
  int bond = tid / (HB_C / 8);
  int h0 = (tid - bond * (HB_C / 8)) * 8;
  union { u32x4 u; __bf16 h[8]; } o;
  if (bond < N_BONDS_C) {
    int a  = b2a[bond];
    int rb = b2revb[bond];
    union { u32x4 u; __bf16 h[8]; } x, y;
    x.u = *(const u32x4*)(amsg + (size_t)a * HB_C + h0);
    y.u = *(const u32x4*)(msg + (size_t)rb * HB_C + h0);
#pragma unroll
    for (int j = 0; j < 8; ++j) o.h[j] = (__bf16)((float)x.h[j] - (float)y.h[j]);
  } else {
#pragma unroll
    for (int j = 0; j < 8; ++j) o.h[j] = (__bf16)0.f;
  }
  *(u32x4*)(t + (size_t)bond * HB_C + h0) = o.u;
}

// ain[atom] = concat(f_atoms[atom] (133), amsg[atom] (300), zeros) padded to 448, bf16
__global__ void build_ain_kernel(const float* __restrict__ f_atoms, const __bf16* __restrict__ amsg,
                                 __bf16* __restrict__ ain) {
  int tid = blockIdx.x * blockDim.x + threadIdx.x;
  constexpr int tot = MA_C * (KAI_C / 8);
  if (tid >= tot) return;
  int atom = tid / (KAI_C / 8);
  int h0 = (tid - atom * (KAI_C / 8)) * 8;
  union { u32x4 u; __bf16 h[8]; } o;
#pragma unroll
  for (int j = 0; j < 8; ++j) {
    int col = h0 + j;
    float v = 0.f;
    if (atom < N_ATOMS_C) {
      if (col < AFD_C)            v = f_atoms[(size_t)atom * AFD_C + col];
      else if (col < AFD_C + H_C) v = (float)amsg[(size_t)atom * HB_C + (col - AFD_C)];
    }
    o.h[j] = (__bf16)v;
  }
  *(u32x4*)(ain + (size_t)atom * KAI_C + h0) = o.u;
}

// Per-molecule mean (deterministic: sorted segment ranges via binary search, no atomics)
__global__ void mol_mean_kernel(const float* __restrict__ ah, const int* __restrict__ mol_id,
                                float* __restrict__ out) {
  __shared__ int se[2];
  const int m = blockIdx.x;
  if (threadIdx.x < 2) {
    int target = m + (int)threadIdx.x;
    int lo = 1, hi = N_ATOMS_C;           // index 0 is the dummy segment, excluded
    while (lo < hi) {
      int mid = (lo + hi) >> 1;
      if (mol_id[mid] < target) lo = mid + 1; else hi = mid;
    }
    se[threadIdx.x] = lo;
  }
  __syncthreads();
  const int h = threadIdx.x;
  if (h >= H_C) return;
  const int s = se[0], e = se[1];
  float sum = 0.f;
  for (int i = s; i < e; ++i) sum += ah[(size_t)i * HB_C + h];
  out[(size_t)m * H_C + h] = (e > s) ? (sum / (float)(e - s)) : 0.f;
}

// ---------------- host launcher ----------------
extern "C" void kernel_launch(void* const* d_in, const int* in_sizes, int n_in,
                              void* d_out, int out_size, void* d_ws, size_t ws_size,
                              hipStream_t stream) {
  (void)in_sizes; (void)n_in; (void)out_size;
  const float* f_atoms = (const float*)d_in[0];
  const float* f_bonds = (const float*)d_in[1];
  const int*   a2b     = (const int*)d_in[2];
  const int*   b2a     = (const int*)d_in[3];
  const int*   b2revb  = (const int*)d_in[4];
  const int*   mol_id  = (const int*)d_in[5];
  const float* W_i     = (const float*)d_in[6];
  const float* W_h     = (const float*)d_in[7];
  const float* W_o     = (const float*)d_in[8];
  const float* b_o     = (const float*)d_in[9];
  float* out = (float*)d_out;

  char* ws = (char*)d_ws;
  size_t off = 0;
  auto alloc = [&](size_t bytes) { size_t r = off; off += (bytes + 255) & ~(size_t)255; return r; };

  const size_t o_inp  = alloc((size_t)MB_C * HB_C * 2);
  const size_t o_msg  = alloc((size_t)MB_C * HB_C * 2);
  const size_t o_t    = alloc((size_t)MB_C * HB_C * 2);
  const size_t o_amsg = alloc((size_t)MA_C * HB_C * 2);
  const size_t o_ain  = alloc((size_t)MA_C * KAI_C * 2);   // fb (MB*160*2) overlaps here
  const size_t o_ah   = alloc((size_t)MA_C * HB_C * 4);
  const size_t o_wi   = alloc((size_t)(KFB_C / 32) * NTILES_C * 32 * 16 * 2);
  const size_t o_wh   = alloc((size_t)(HB_C / 32) * NTILES_C * 32 * 16 * 2);
  const size_t o_wo   = alloc((size_t)(KAI_C / 32) * NTILES_C * 32 * 16 * 2);
  if (ws_size < off) return;   // workspace too small; nothing safe to do

  __bf16* inp  = (__bf16*)(ws + o_inp);
  __bf16* msg  = (__bf16*)(ws + o_msg);
  __bf16* tbuf = (__bf16*)(ws + o_t);
  __bf16* amsg = (__bf16*)(ws + o_amsg);
  __bf16* ain  = (__bf16*)(ws + o_ain);
  __bf16* fb   = (__bf16*)(ws + o_ain);   // used only before ain is needed
  float*  ah   = (float*)(ws + o_ah);
  __bf16* wiP  = (__bf16*)(ws + o_wi);
  __bf16* whP  = (__bf16*)(ws + o_wh);
  __bf16* woP  = (__bf16*)(ws + o_wo);

  const int T = 256;
  // prep
  {
    int tot = MB_C * KFB_C;
    conv_bonds_kernel<<<(tot + T - 1) / T, T, 0, stream>>>(f_bonds, fb);
  }
  pack_w_kernel<<<((KFB_C / 32) * NTILES_C * 32 + T - 1) / T, T, 0, stream>>>(W_i, BFD_C, H_C, KFB_C, wiP);
  pack_w_kernel<<<((HB_C / 32) * NTILES_C * 32 + T - 1) / T, T, 0, stream>>>(W_h, H_C, H_C, HB_C, whP);
  pack_w_kernel<<<((KAI_C / 32) * NTILES_C * 32 + T - 1) / T, T, 0, stream>>>(W_o, AFD_C + H_C, H_C, KAI_C, woP);

  const int MtB = MB_C / 16, MtA = MA_C / 16;
  const int gB = (MtB * 2 + 3) / 4;
  const int gA = (MtA * 2 + 3) / 4;

  // inp = fb @ W_i ; msg = relu(inp)
  gemm_wi_kernel<<<gB, 128, 0, stream>>>(fb, wiP, inp, msg, MtB, KFB_C / 32, KFB_C);

  const int aggB = (MA_C * (HB_C / 8) + T - 1) / T;
  const int tB   = (MB_C * (HB_C / 8) + T - 1) / T;
  for (int it = 0; it < 2; ++it) {   // DEPTH-1 message passing steps
    agg_atoms_kernel<<<aggB, T, 0, stream>>>(msg, a2b, amsg);
    build_t_kernel<<<tB, T, 0, stream>>>(amsg, msg, b2a, b2revb, tbuf);
    gemm_wh_kernel<<<gB, 128, 0, stream>>>(tbuf, whP, inp, msg, MtB, HB_C / 32, HB_C);
  }

  // readout
  agg_atoms_kernel<<<aggB, T, 0, stream>>>(msg, a2b, amsg);
  {
    int tot = MA_C * (KAI_C / 8);
    build_ain_kernel<<<(tot + T - 1) / T, T, 0, stream>>>(f_atoms, amsg, ain);
  }
  gemm_wo_kernel<<<gA, 128, 0, stream>>>(ain, woP, b_o, ah, MtA, KAI_C / 32, KAI_C);
  mol_mean_kernel<<<N_MOLS_C, HB_C, 0, stream>>>(ah, mol_id, out);
}
